// LPCNet_51213190037736
// MI455X (gfx1250) — compile-verified, matrix operand order checked by reading, not run on hardware
//
#include <hip/hip_runtime.h>
#include <hip/hip_bf16.h>

typedef __attribute__((ext_vector_type(16))) _Float16 v16h;
typedef __attribute__((ext_vector_type(8)))  _Float16 v8h;
typedef __attribute__((ext_vector_type(8)))  float    v8f;

#define FRAME 160
#define BS 16
#define CS 2400
#define ROWS (BS*CS)   /* 38400 */
#define NF 15
#define H1 384
#define G1 1152
#define H2 16
#define G2 48
#define KIN 512

static __device__ __forceinline__ float sigmoidf_(float x){ return 1.f/(1.f+__expf(-x)); }

// Load a 16x32 f16 A-fragment (CDNA5 layout): lanes 0-15 = rows M0..15 (K-run kh*8..+7
// in elems 0..7, 16+kh*8..+7 in elems 8..15), lanes 16-31 same rows with kh=1.
static __device__ __forceinline__ v16h load_frag(const _Float16* base, int ld, int lane){
  int r  = lane & 15;
  int kh = (lane >> 4) & 1;
  const _Float16* p = base + r*ld + kh*8;
  v16h a;
  *((v8h*)&a)       = *((const v8h*)p);
  *(((v8h*)&a) + 1) = *((const v8h*)(p + 16));
  return a;
}

// ---------------- small helpers ----------------
__global__ void k_f32tof16(const float* __restrict__ s, _Float16* __restrict__ d, int n){
  int i = blockIdx.x*256 + threadIdx.x;
  if (i < n) d[i] = (_Float16)s[i];
}
__global__ void k_fcpad(const float* __restrict__ s /*[256,16]*/, _Float16* __restrict__ d /*[256,32]*/){
  int i = blockIdx.x*256 + threadIdx.x;            // 0..8191
  int o = i >> 5, k = i & 31;
  d[i] = (k < 16) ? (_Float16)s[o*16 + k] : (_Float16)0.f;
}

// ---------------- encoder: conv1d x2 + dense x2 -> f16 features [16][15][128] -------------
__global__ __launch_bounds__(128) void k_encoder(
    const float* __restrict__ feat, const int* __restrict__ pitch,
    const float* __restrict__ emb_pitch,
    const float* __restrict__ w1, const float* __restrict__ b1,
    const float* __restrict__ w2, const float* __restrict__ b2,
    const float* __restrict__ d1w, const float* __restrict__ d1b,
    const float* __restrict__ d2w, const float* __restrict__ d2b,
    _Float16* __restrict__ fout)
{
  __shared__ float xin[19*102];
  __shared__ float h1s[17*128];
  __shared__ float h2s[15*128];
  __shared__ float h3s[15*128];
  int b = blockIdx.x, tid = threadIdx.x;
  for (int i = tid; i < 19*102; i += 128){
    int fr = i/102, c = i%102;
    float v;
    if (c < 38) v = feat[(b*19+fr)*38 + c];
    else {
      int pi = pitch[b*19 + fr];
      v = emb_pitch[pi*64 + (c-38)];
    }
    xin[i] = v;
  }
  __syncthreads();
  int o = tid;
  for (int p = 0; p < 17; ++p){
    float acc = b1[o];
    for (int i = 0; i < 102; ++i){
      const float* wr = w1 + (o*102 + i)*3;
      acc += xin[(p+0)*102+i]*wr[0] + xin[(p+1)*102+i]*wr[1] + xin[(p+2)*102+i]*wr[2];
    }
    h1s[p*128+o] = tanhf(acc);
  }
  __syncthreads();
  for (int p = 0; p < 15; ++p){
    float acc = b2[o];
    for (int i = 0; i < 128; ++i){
      const float* wr = w2 + (o*128 + i)*3;
      acc += h1s[(p+0)*128+i]*wr[0] + h1s[(p+1)*128+i]*wr[1] + h1s[(p+2)*128+i]*wr[2];
    }
    h2s[p*128+o] = tanhf(acc);
  }
  __syncthreads();
  for (int p = 0; p < 15; ++p){
    float acc = d1b[o];
    for (int i = 0; i < 128; ++i) acc += h2s[p*128+i]*d1w[o*128+i];
    h3s[p*128+o] = tanhf(acc);
  }
  __syncthreads();
  for (int p = 0; p < 15; ++p){
    float acc = d2b[o];
    for (int i = 0; i < 128; ++i) acc += h3s[p*128+i]*d2w[o*128+i];
    fout[(b*15+p)*128 + o] = (_Float16)tanhf(acc);
  }
}

// ---------------- build X1 (emb_pcm||rfeat) f16 and X2's rfeat tail ----------------
__global__ __launch_bounds__(128) void k_build_x1(
    const int* __restrict__ pcm,        // [16,2400,3]
    const float* __restrict__ emb_pcm,  // [256,128]
    const _Float16* __restrict__ F16f,  // [16,15,128]
    _Float16* __restrict__ X1, _Float16* __restrict__ X2)
{
  int t = blockIdx.x, b = blockIdx.y, c = threadIdx.x;
  int row = t*16 + b;
  const int* pp = pcm + (size_t)(b*CS + t)*3;
  _Float16* x1 = X1 + (size_t)row*KIN;
  #pragma unroll
  for (int s = 0; s < 3; ++s){
    int pi = pp[s];
    x1[s*128 + c] = (_Float16)emb_pcm[pi*128 + c];
  }
  _Float16 fv = F16f[(b*NF + (t/FRAME))*128 + c];
  x1[384 + c] = fv;
  X2[(size_t)row*KIN + 384 + c] = fv;
}

// ---------------- WMMA GEMM: Y[M,N] = X[M,512] * W[N,512]^T + bias ----------------
__global__ __launch_bounds__(256) void k_gemm512(
    const _Float16* __restrict__ X,
    const _Float16* __restrict__ W,
    const float*    __restrict__ bias,
    float*          __restrict__ Y,
    int N)
{
  __shared__ _Float16 As[16*520];          // padded stride -> conflict-free b128 ds loads
  int tid = threadIdx.x;
  int r0 = blockIdx.x * 16;
  const _Float16* xs = X + (size_t)r0*KIN;
  for (int c = tid; c < 16*64; c += 256){  // 64 x v8h chunks per row
    int row = c >> 6, k8 = (c & 63) << 3;
    *((v8h*)&As[row*520 + k8]) = *((const v8h*)(xs + row*KIN + k8));
  }
  __syncthreads();
  int wave = tid >> 5, lane = tid & 31;
  int nt = blockIdx.y*8 + wave;
  if (nt*16 < N){
    int n0 = nt*16;
    int kh = (lane >> 4) & 1;
    const _Float16* wp = W + (size_t)(n0 + (lane & 15))*KIN + kh*8;
    v8f acc = {};
    for (int kb = 0; kb < KIN; kb += 32){
      v16h a = load_frag(&As[kb], 520, lane);
      v16h bm;
      *((v8h*)&bm)       = *((const v8h*)(wp + kb));
      *(((v8h*)&bm) + 1) = *((const v8h*)(wp + kb + 16));
      acc = __builtin_amdgcn_wmma_f32_16x16x32_f16(false, a, false, bm, (short)0, acc, false, false);
    }
    int col = n0 + (lane & 15);
    float bv = bias[col];
    float* yp = Y + (size_t)r0*N + col;
    #pragma unroll
    for (int v = 0; v < 8; ++v){
      int row = v + 8*kh;
      yp[(size_t)row*N] = acc[v] + bv;
    }
  }
}

// ---------------- GRU1 scan: 2400 steps, WMMA recurrence (M=batch=16) ----------------
__global__ __launch_bounds__(1024) void k_scan1(
    const float*    __restrict__ GX,   // [ROWS,1152], row = t*16+b (bih already added)
    const _Float16* __restrict__ Whh,  // [1152,384] f16
    const float*    __restrict__ bhh,  // [1152]
    _Float16*       __restrict__ X2)   // [ROWS,512]; writes cols 0..383 = h_t (f16)
{
  extern __shared__ char smem[];
  _Float16* Hh = (_Float16*)smem;            // [16][392] f16, padded stride
  float*    GH = (float*)(smem + 16*392*2);  // [16][1160] f32, padded stride
  int tid = threadIdx.x, lane = tid & 31, wave = tid >> 5;
  for (int i = tid; i < 16*392; i += 1024) Hh[i] = (_Float16)0.f;
  int bidx[6], jidx[6];
  float hreg[6];
  #pragma unroll
  for (int k = 0; k < 6; ++k){
    int idx = tid + k*1024;                  // 6144 = 16*384 items
    bidx[k] = idx / 384; jidx[k] = idx % 384;
    hreg[k] = 0.f;
  }
  __syncthreads();
  for (int t = 0; t < CS; ++t){
    // phase A: GH = Hh @ Whh^T + bhh   (16 x 1152, K=384) -- 72 tiles over 32 waves
    for (int nt = wave; nt < 72; nt += 32){
      int n0 = nt*16;
      int kh = (lane >> 4) & 1;
      const _Float16* wp = Whh + (size_t)(n0 + (lane & 15))*H1 + kh*8;
      v8f acc = {};
      #pragma unroll
      for (int kb = 0; kb < H1; kb += 32){
        v16h a = load_frag(&Hh[kb], 392, lane);
        v16h bm;
        *((v8h*)&bm)       = *((const v8h*)(wp + kb));
        *(((v8h*)&bm) + 1) = *((const v8h*)(wp + kb + 16));
        acc = __builtin_amdgcn_wmma_f32_16x16x32_f16(false, a, false, bm, (short)0, acc, false, false);
      }
      int col = n0 + (lane & 15);
      float bv = bhh[col];
      #pragma unroll
      for (int v = 0; v < 8; ++v){
        int bb = v + 8*kh;                   // batch row of D tile
        GH[bb*1160 + col] = acc[v] + bv;
      }
    }
    __syncthreads();
    // phase B: gate nonlinearities + h update (torch gate order r,z,n)
    const float* gxt = GX + (size_t)t*16*G1;
    #pragma unroll
    for (int k = 0; k < 6; ++k){
      int b = bidx[k], j = jidx[k];
      const float* gx = gxt + b*G1;
      float r = sigmoidf_(gx[j]       + GH[b*1160 + j]);
      float z = sigmoidf_(gx[384 + j] + GH[b*1160 + 384 + j]);
      float n = tanhf(gx[768 + j] + r * GH[b*1160 + 768 + j]);
      float h = (1.f - z)*n + z*hreg[k];
      hreg[k] = h;
      Hh[b*392 + j] = (_Float16)h;
      X2[(size_t)(t*16 + b)*KIN + j] = (_Float16)h;
    }
    __syncthreads();
  }
}

// ---------------- GRU2 scan: hidden=16, VALU ----------------
__global__ __launch_bounds__(256) void k_scan2(
    const float* __restrict__ GX,    // [ROWS,48]
    const float* __restrict__ whh,   // [48,16]
    const float* __restrict__ bhh,   // [48]
    _Float16*    __restrict__ OUTP)  // [ROWS,32] (cols 16..31 zero pad)
{
  __shared__ float H[16*17];
  __shared__ float Wl[48*16];
  __shared__ float Bl[48];
  int tid = threadIdx.x;
  int b = tid >> 4, j = tid & 15;
  for (int i = tid; i < 48*16; i += 256) Wl[i] = whh[i];
  if (tid < 48) Bl[tid] = bhh[tid];
  H[b*17 + j] = 0.f;
  float h = 0.f;
  __syncthreads();
  for (int t = 0; t < CS; ++t){
    float hr = Bl[j], hz = Bl[16 + j], hn = Bl[32 + j];
    #pragma unroll
    for (int i = 0; i < 16; ++i){
      float hv = H[b*17 + i];
      hr += Wl[j*16 + i]*hv;
      hz += Wl[(16 + j)*16 + i]*hv;
      hn += Wl[(32 + j)*16 + i]*hv;
    }
    const float* gx = GX + (size_t)(t*16 + b)*G2;
    float r = sigmoidf_(gx[j] + hr);
    float z = sigmoidf_(gx[16 + j] + hz);
    float n = tanhf(gx[32 + j] + r*hn);
    h = (1.f - z)*n + z*h;
    __syncthreads();
    H[b*17 + j] = h;
    _Float16* op = OUTP + (size_t)(t*16 + b)*32;
    op[j]      = (_Float16)h;
    op[16 + j] = (_Float16)0.f;
    __syncthreads();
  }
}

// ---------------- MDense head: 2 WMMA (K padded to 32) + tanh/gamma epilogue ----------
__global__ __launch_bounds__(256) void k_head(
    const _Float16* __restrict__ OUTP,  // [ROWS,32]
    const _Float16* __restrict__ F1,    // [256,32] padded
    const _Float16* __restrict__ F2,    // [256,32] padded
    const float* __restrict__ b1, const float* __restrict__ b2,
    const float* __restrict__ g1, const float* __restrict__ g2,
    float* __restrict__ out)            // [16][2400][256]
{
  int t = blockIdx.x;
  int wave = threadIdx.x >> 5, lane = threadIdx.x & 31;
  int c0 = (blockIdx.y*8 + wave)*16;
  int kh = (lane >> 4) & 1;
  v16h a = load_frag(OUTP + (size_t)t*16*32, 32, lane);
  v16h w1, w2;
  const _Float16* p1 = F1 + (size_t)(c0 + (lane & 15))*32 + kh*8;
  *((v8h*)&w1)       = *((const v8h*)p1);
  *(((v8h*)&w1) + 1) = *((const v8h*)(p1 + 16));
  const _Float16* p2 = F2 + (size_t)(c0 + (lane & 15))*32 + kh*8;
  *((v8h*)&w2)       = *((const v8h*)p2);
  *(((v8h*)&w2) + 1) = *((const v8h*)(p2 + 16));
  v8f y1 = {}, y2 = {};
  y1 = __builtin_amdgcn_wmma_f32_16x16x32_f16(false, a, false, w1, (short)0, y1, false, false);
  y2 = __builtin_amdgcn_wmma_f32_16x16x32_f16(false, a, false, w2, (short)0, y2, false, false);
  int col = c0 + (lane & 15);
  float bb1 = b1[col], bb2 = b2[col], gg1 = g1[col], gg2 = g2[col];
  #pragma unroll
  for (int v = 0; v < 8; ++v){
    int bbat = v + 8*kh;                 // batch index (row t*16+bbat)
    float p = gg1*tanhf(y1[v] + bb1) + gg2*tanhf(y2[v] + bb2);
    out[((size_t)bbat*CS + t)*256 + col] = p;
  }
}

// ---------------- host ----------------
extern "C" void kernel_launch(void* const* d_in, const int* in_sizes, int n_in,
                              void* d_out, int out_size, void* d_ws, size_t ws_size,
                              hipStream_t stream)
{
  (void)in_sizes; (void)n_in; (void)out_size; (void)ws_size;
  const int*   pcm       = (const int*)  d_in[0];
  const float* feat      = (const float*)d_in[1];
  const int*   pitch     = (const int*)  d_in[2];
  const float* emb_pcm   = (const float*)d_in[3];
  const float* emb_pitch = (const float*)d_in[4];
  const float* conv1_w   = (const float*)d_in[5];
  const float* conv1_b   = (const float*)d_in[6];
  const float* conv2_w   = (const float*)d_in[7];
  const float* conv2_b   = (const float*)d_in[8];
  const float* d1_w  = (const float*)d_in[9];
  const float* d1_b  = (const float*)d_in[10];
  const float* d2_w  = (const float*)d_in[11];
  const float* d2_b  = (const float*)d_in[12];
  const float* g1_wih = (const float*)d_in[13];
  const float* g1_whh = (const float*)d_in[14];
  const float* g1_bih = (const float*)d_in[15];
  const float* g1_bhh = (const float*)d_in[16];
  const float* g2_wih = (const float*)d_in[17];
  const float* g2_whh = (const float*)d_in[18];
  const float* g2_bih = (const float*)d_in[19];
  const float* g2_bhh = (const float*)d_in[20];
  const float* fc1_w = (const float*)d_in[21];
  const float* fc1_b = (const float*)d_in[22];
  const float* fc2_w = (const float*)d_in[23];
  const float* fc2_b = (const float*)d_in[24];
  const float* gamma1 = (const float*)d_in[25];
  const float* gamma2 = (const float*)d_in[26];

  char* ws = (char*)d_ws; size_t off = 0;
  auto alloc = [&](size_t bytes)->char* {
    char* p = ws + off; off += (bytes + 255) & ~(size_t)255; return p;
  };
  _Float16* W1h  = (_Float16*)alloc((size_t)G1*KIN*2);
  _Float16* Whh1 = (_Float16*)alloc((size_t)G1*H1*2);
  _Float16* W2h  = (_Float16*)alloc((size_t)G2*KIN*2);
  _Float16* F1p  = (_Float16*)alloc(256*32*2);
  _Float16* F2p  = (_Float16*)alloc(256*32*2);
  _Float16* F16f = (_Float16*)alloc((size_t)BS*NF*128*2);
  _Float16* X1   = (_Float16*)alloc((size_t)ROWS*KIN*2);
  _Float16* X2   = (_Float16*)alloc((size_t)ROWS*KIN*2);
  float*    GX1  = (float*)   alloc((size_t)ROWS*G1*4);
  float*    GX2  = (float*)   alloc((size_t)ROWS*G2*4);
  _Float16* O2P  = (_Float16*)alloc((size_t)ROWS*32*2);

  k_f32tof16<<<(G1*KIN + 255)/256, 256, 0, stream>>>(g1_wih, W1h, G1*KIN);
  k_f32tof16<<<(G1*H1 + 255)/256, 256, 0, stream>>>(g1_whh, Whh1, G1*H1);
  k_f32tof16<<<(G2*KIN + 255)/256, 256, 0, stream>>>(g2_wih, W2h, G2*KIN);
  k_fcpad<<<32, 256, 0, stream>>>(fc1_w, F1p);
  k_fcpad<<<32, 256, 0, stream>>>(fc2_w, F2p);

  k_encoder<<<BS, 128, 0, stream>>>(feat, pitch, emb_pitch, conv1_w, conv1_b,
                                    conv2_w, conv2_b, d1_w, d1_b, d2_w, d2_b, F16f);
  k_build_x1<<<dim3(CS, BS), 128, 0, stream>>>(pcm, emb_pcm, F16f, X1, X2);

  k_gemm512<<<dim3(ROWS/16, 9), 256, 0, stream>>>(X1, W1h, g1_bih, GX1, G1);

  size_t smem1 = (size_t)16*392*2 + (size_t)16*1160*4;   // 86,784 B
  k_scan1<<<1, 1024, smem1, stream>>>(GX1, Whh1, g1_bhh, X2);

  k_gemm512<<<dim3(ROWS/16, 1), 256, 0, stream>>>(X2, W2h, g2_bih, GX2, G2);
  k_scan2<<<1, 256, 0, stream>>>(GX2, g2_whh, g2_bhh, O2P);

  k_head<<<dim3(CS, 2), 256, 0, stream>>>(O2P, F1p, F2p, fc1_b, fc2_b, gamma1, gamma2,
                                          (float*)d_out);
}
